// Model_22728966930623
// MI455X (gfx1250) — compile-verified, matrix operand order checked by reading, not run on hardware
//
#include <hip/hip_runtime.h>
#include <math.h>

// ---------------- model constants ----------------
#define BB   2
#define SS   4096
#define DD   768
#define HH   12
#define DHH  64
#define LL   4
#define FFD  3072
#define WW   256
#define NBK  16          // S / W
#define NTOK (BB*SS)     // 8192

typedef __attribute__((ext_vector_type(16))) __bf16 v16bf;
typedef __attribute__((ext_vector_type(8)))  float  v8f;

// float -> bf16 (round to nearest even)
__device__ inline __bf16 f2bf(float f) {
    union { float f; unsigned u; } c; c.f = f;
    unsigned r = (c.u + 0x7FFFu + ((c.u >> 16) & 1u)) >> 16;
    union { unsigned short s; __bf16 b; } o; o.s = (unsigned short)r;
    return o.b;
}

__device__ inline v8f wmma_bf16(v16bf a, v16bf b, v8f c) {
    return __builtin_amdgcn_wmma_f32_16x16x32_bf16(
        false, a, false, b, (short)0, c, false, false);
}

// Vectorized 16x32 bf16 fragment load from LDS stored row-major-along-K.
// Lane (row = row0 + lane&15, khalf = lane>>4) needs two contiguous
// 8-element runs at k = kh and k = kh+16  ->  2x ds_load_b128.
// Requires ld % 8 == 0 and k0 % 8 == 0.
__device__ inline v16bf frag_ld(const __bf16* s, int ld, int row0, int k0, int lane) {
    int r  = row0 + (lane & 15);
    int kh = ((lane >> 4) & 1) << 3;
    const uint4* p = (const uint4*)(s + r * ld + k0 + kh);
    union { uint4 u[2]; v16bf f; } c;
    c.u[0] = p[0];
    c.u[1] = p[2];
    return c.f;
}

__device__ inline float block_sum(float v, float* red) {
    int tid = threadIdx.x;
    red[tid] = v; __syncthreads();
    for (int o = 128; o > 0; o >>= 1) {
        if (tid < o) red[tid] += red[tid + o];
        __syncthreads();
    }
    float r = red[0]; __syncthreads();
    return r;
}

// ------------- weight convert+transpose: f32 [K][N] -> bf16 [N][K] -----------
__global__ void __launch_bounds__(256)
transpose_convert_kernel(const float* __restrict__ src, __bf16* __restrict__ dst,
                         int K, int N) {
    __shared__ float tile[32][33];
    size_t off = (size_t)blockIdx.z * K * N;
    int k0 = blockIdx.x * 32, n0 = blockIdx.y * 32;
    int tx = threadIdx.x & 31, ty = threadIdx.x >> 5;   // ty = 0..7
    #pragma unroll
    for (int i = 0; i < 4; ++i)
        tile[ty + i * 8][tx] = src[off + (size_t)(k0 + ty + i * 8) * N + n0 + tx];
    __syncthreads();
    #pragma unroll
    for (int i = 0; i < 4; ++i)
        dst[off + (size_t)(n0 + ty + i * 8) * K + k0 + tx] = f2bf(tile[tx][ty + i * 8]);
}

// ---------------- embedding + LayerNorm (f32 + bf16 outputs) ----------------
__global__ void __launch_bounds__(256)
embed_ln_kernel(const int* __restrict__ ids,
                const float* __restrict__ wemb, const float* __restrict__ pemb,
                const float* __restrict__ g, const float* __restrict__ bta,
                float* __restrict__ outF, __bf16* __restrict__ outB) {
    __shared__ float red[256];
    size_t t = blockIdx.x;
    int s = (int)(t % SS);
    int tid = threadIdx.x;
    int id = ids[t];
    float v[3]; float sum = 0.f;
    #pragma unroll
    for (int i = 0; i < 3; ++i) {
        int c = tid + i * 256;
        float val = wemb[(size_t)id * DD + c] + pemb[(size_t)s * DD + c];
        v[i] = val; sum += val;
    }
    float mean = block_sum(sum, red) * (1.f / DD);
    float sq = 0.f;
    #pragma unroll
    for (int i = 0; i < 3; ++i) { float d = v[i] - mean; sq += d * d; }
    float var = block_sum(sq, red) * (1.f / DD);
    float rstd = rsqrtf(var + 1e-5f);
    #pragma unroll
    for (int i = 0; i < 3; ++i) {
        int c = tid + i * 256;
        float o = (v[i] - mean) * rstd * g[c] + bta[c];
        outF[t * DD + c] = o;
        outB[t * DD + c] = f2bf(o);
    }
}

// ---------------- residual + LayerNorm (f32 + bf16 outputs) ----------------
__global__ void __launch_bounds__(256)
add_ln_kernel(const float* __restrict__ x, const float* __restrict__ r,
              const float* __restrict__ g, const float* __restrict__ bta,
              float* __restrict__ outF, __bf16* __restrict__ outB) {
    __shared__ float red[256];
    size_t t = blockIdx.x;
    int tid = threadIdx.x;
    float v[3]; float sum = 0.f;
    #pragma unroll
    for (int i = 0; i < 3; ++i) {
        int c = tid + i * 256;
        float val = x[t * DD + c] + r[t * DD + c];
        v[i] = val; sum += val;
    }
    float mean = block_sum(sum, red) * (1.f / DD);
    float sq = 0.f;
    #pragma unroll
    for (int i = 0; i < 3; ++i) { float d = v[i] - mean; sq += d * d; }
    float var = block_sum(sq, red) * (1.f / DD);
    float rstd = rsqrtf(var + 1e-5f);
    #pragma unroll
    for (int i = 0; i < 3; ++i) {
        int c = tid + i * 256;
        float o = (v[i] - mean) * rstd * g[c] + bta[c];
        outF[t * DD + c] = o;
        outB[t * DD + c] = f2bf(o);
    }
}

// ---------------- bf16 WMMA GEMM ----------------
// Out[M,N] = A[M,K] @ W[K,N] + bias; A bf16 [M][K], W pre-transposed bf16 [N][K].
// Block tile 128x128, BK=32, 256 thr = 8 waves (2x4), wave tile 64x32.
// Double-buffered LDS; split staging: global->regs before compute,
// regs->LDS after compute (keeps wmma's dscnt wait off the stores).
#define GLDA 40
__global__ void __launch_bounds__(256)
gemm_bias_kernel(const __bf16* __restrict__ A, const __bf16* __restrict__ Bt,
                 const float* __restrict__ bias,
                 float* __restrict__ outF, __bf16* __restrict__ outB,
                 __bf16* __restrict__ outBT,   // optional [b][h][dh][s] bf16
                 int N, int Kd, int act) {
    __shared__ __bf16 sA[2][128 * GLDA];
    __shared__ __bf16 sB[2][128 * GLDA];
    int tid = threadIdx.x, wave = tid >> 5, lane = tid & 31;
    int wm = wave >> 2, wn = wave & 3;
    int gm0 = blockIdx.x * 128, gn0 = blockIdx.y * 128;

    v8f acc[4][2];
    #pragma unroll
    for (int i = 0; i < 4; ++i)
        #pragma unroll
        for (int j = 0; j < 2; ++j)
            #pragma unroll
            for (int e = 0; e < 8; ++e) acc[i][j][e] = 0.f;

    int r0 = tid >> 2, r1 = (tid + 256) >> 2;       // staged rows (0..127)
    int c0 = (tid & 3) * 8, c1 = ((tid + 256) & 3) * 8;

    int nk = Kd >> 5;
    uint4 ra0, ra1, rb0, rb1;
    // prologue: stage tile 0
    ra0 = *(const uint4*)(A  + (size_t)(gm0 + r0) * Kd + c0);
    ra1 = *(const uint4*)(A  + (size_t)(gm0 + r1) * Kd + c1);
    rb0 = *(const uint4*)(Bt + (size_t)(gn0 + r0) * Kd + c0);
    rb1 = *(const uint4*)(Bt + (size_t)(gn0 + r1) * Kd + c1);
    *(uint4*)(sA[0] + r0 * GLDA + c0) = ra0;
    *(uint4*)(sA[0] + r1 * GLDA + c1) = ra1;
    *(uint4*)(sB[0] + r0 * GLDA + c0) = rb0;
    *(uint4*)(sB[0] + r1 * GLDA + c1) = rb1;
    __syncthreads();

    for (int kt = 0; kt < nk; ++kt) {
        int cur = kt & 1;
        bool pf = (kt + 1 < nk);
        if (pf) {                                   // global loads issue early
            int k0 = (kt + 1) << 5;
            ra0 = *(const uint4*)(A  + (size_t)(gm0 + r0) * Kd + k0 + c0);
            ra1 = *(const uint4*)(A  + (size_t)(gm0 + r1) * Kd + k0 + c1);
            rb0 = *(const uint4*)(Bt + (size_t)(gn0 + r0) * Kd + k0 + c0);
            rb1 = *(const uint4*)(Bt + (size_t)(gn0 + r1) * Kd + k0 + c1);
        }
        v16bf bf0 = frag_ld(sB[cur], GLDA, wn * 32,      0, lane);
        v16bf bf1 = frag_ld(sB[cur], GLDA, wn * 32 + 16, 0, lane);
        #pragma unroll
        for (int ms = 0; ms < 4; ++ms) {
            v16bf af = frag_ld(sA[cur], GLDA, wm * 64 + ms * 16, 0, lane);
            acc[ms][0] = wmma_bf16(af, bf0, acc[ms][0]);
            acc[ms][1] = wmma_bf16(af, bf1, acc[ms][1]);
        }
        if (pf) {                                   // LDS stores after compute
            __bf16* dA = sA[cur ^ 1]; __bf16* dB = sB[cur ^ 1];
            *(uint4*)(dA + r0 * GLDA + c0) = ra0;
            *(uint4*)(dA + r1 * GLDA + c1) = ra1;
            *(uint4*)(dB + r0 * GLDA + c0) = rb0;
            *(uint4*)(dB + r1 * GLDA + c1) = rb1;
        }
        __syncthreads();
    }
    int khalf = (lane >> 4) & 1;
    #pragma unroll
    for (int j = 0; j < 2; ++j) {
        int nn = gn0 + wn * 32 + j * 16 + (lane & 15);
        float bv = bias[nn];
        #pragma unroll
        for (int ms = 0; ms < 4; ++ms)
            #pragma unroll
            for (int e = 0; e < 8; ++e) {
                int m = gm0 + wm * 64 + ms * 16 + e + 8 * khalf;
                float v = acc[ms][j][e] + bv;
                if (act == 1) v = 0.5f * v * (1.f + erff(v * 0.70710678118f));
                if (outF) outF[(size_t)m * N + nn] = v;
                if (outB) outB[(size_t)m * N + nn] = f2bf(v);
                if (outBT) {
                    int bb = m >> 12, s = m & (SS - 1);
                    int h = nn >> 6, dh = nn & 63;
                    outBT[(((size_t)bb * HH + h) * DHH + dh) * SS + s] = f2bf(v);
                }
            }
    }
}

// ---------------- banded attention (flash-style, WMMA, bf16 in/out) ----------
// One WG per (b, h, block nb, q-chunk of 64 rows); 12 key tiles of 64.
// V arrives pre-transposed [b][h][dh][s] -> all tile copies are b128.
#define ALD 72
__global__ void __launch_bounds__(256)
band_attn_kernel(const __bf16* __restrict__ Q, const __bf16* __restrict__ Km,
                 const __bf16* __restrict__ Vt, const int* __restrict__ mask,
                 __bf16* __restrict__ Out) {
    __shared__ __bf16 sQ[64 * ALD];    // [qrow][dh]
    __shared__ __bf16 sK[64 * ALD];    // [kpos][dh]
    __shared__ __bf16 sVt[64 * ALD];   // [dh][kpos]
    __shared__ __bf16 sP[64 * ALD];    // [qrow][kpos]
    __shared__ float  sS[64 * 65];
    __shared__ float  mI[64], lI[64], aL[64];
    __shared__ float  sRed[256];

    int bid = blockIdx.x;
    int qc = bid & 3;
    int nb = (bid >> 2) % NBK;
    int hh = ((bid >> 2) / NBK) % HH;
    int b  = (bid >> 2) / (NBK * HH);
    int tid = threadIdx.x, wave = tid >> 5, lane = tid & 31;
    int qbase = nb * WW + qc * 64;

    int rQ = tid >> 3, cQ = (tid & 7) * 8;          // 64 rows x 8 chunks
    int rQ2 = (tid + 256) >> 3, cQ2 = ((tid + 256) & 7) * 8;

    // Q tile: 64x64 bf16 = 512 16B chunks
    *(uint4*)(sQ + rQ  * ALD + cQ ) =
        *(const uint4*)(Q + ((size_t)b * SS + qbase + rQ ) * DD + hh * 64 + cQ );
    *(uint4*)(sQ + rQ2 * ALD + cQ2) =
        *(const uint4*)(Q + ((size_t)b * SS + qbase + rQ2) * DD + hh * 64 + cQ2);
    if (tid < 64) { mI[tid] = -1e30f; lI[tid] = 0.f; }

    v8f oacc[2];
    #pragma unroll
    for (int i = 0; i < 2; ++i)
        #pragma unroll
        for (int e = 0; e < 8; ++e) oacc[i][e] = 0.f;
    __syncthreads();

    size_t vhead = ((size_t)b * HH + hh) * DHH;     // row base in Vt

    for (int t = 0; t < 12; ++t) {
        int kwin0 = (nb - 1) * WW + t * 64;         // 64-aligned => per-tile validity
        bool tileOK = (kwin0 >= 0) && (kwin0 < SS);
        uint4 z = make_uint4(0, 0, 0, 0);
        uint4 k0v = z, k1v = z, v0v = z, v1v = z;
        if (tileOK) {
            k0v = *(const uint4*)(Km + ((size_t)b * SS + kwin0 + rQ ) * DD + hh * 64 + cQ );
            k1v = *(const uint4*)(Km + ((size_t)b * SS + kwin0 + rQ2) * DD + hh * 64 + cQ2);
            v0v = *(const uint4*)(Vt + (vhead + rQ ) * SS + kwin0 + cQ );
            v1v = *(const uint4*)(Vt + (vhead + rQ2) * SS + kwin0 + cQ2);
        }
        *(uint4*)(sK  + rQ  * ALD + cQ ) = k0v;
        *(uint4*)(sK  + rQ2 * ALD + cQ2) = k1v;
        *(uint4*)(sVt + rQ  * ALD + cQ ) = v0v;
        *(uint4*)(sVt + rQ2 * ALD + cQ2) = v1v;
        __syncthreads();
        // scores: S(64x64) = (Q*K^T)/8
        #pragma unroll
        for (int i = 0; i < 2; ++i) {
            int tile = wave * 2 + i, ms = tile >> 2, ns = tile & 3;
            v8f acc;
            #pragma unroll
            for (int e = 0; e < 8; ++e) acc[e] = 0.f;
            #pragma unroll
            for (int ks = 0; ks < 2; ++ks) {
                v16bf a  = frag_ld(sQ, ALD, ms * 16, ks * 32, lane);
                v16bf bb = frag_ld(sK, ALD, ns * 16, ks * 32, lane);
                acc = wmma_bf16(a, bb, acc);
            }
            int khalf = (lane >> 4) & 1, nn = ns * 16 + (lane & 15);
            #pragma unroll
            for (int e = 0; e < 8; ++e)
                sS[(ms * 16 + e + 8 * khalf) * 65 + nn] = acc[e] * 0.125f;
        }
        __syncthreads();
        // phase A - masked write-back + segment max (4 thr/row)
        {
            int row = tid >> 2, seg = tid & 3, qg = qbase + row;
            float mx = -1e30f;
            #pragma unroll
            for (int j = 0; j < 16; ++j) {
                int c = seg * 16 + j;
                int kg = kwin0 + c;
                int d = kg - qg; if (d < 0) d = -d;
                int kc = kg < 0 ? 0 : (kg >= SS ? SS - 1 : kg);
                bool valid = (kg >= 0) && (kg < SS) && (d <= WW) &&
                             (mask[(size_t)b * SS + kc] > 0);
                float s = valid ? sS[row * 65 + c] : -1e9f;
                sS[row * 65 + c] = s;
                if (s > mx) mx = s;
            }
            sRed[tid] = mx;
        }
        __syncthreads();
        // phase B - per-row max combine, alpha
        if (tid < 64) {
            float tmax = sRed[tid * 4];
            #pragma unroll
            for (int j = 1; j < 4; ++j) { float v = sRed[tid * 4 + j]; if (v > tmax) tmax = v; }
            float mOld = mI[tid];
            float mNew = mOld > tmax ? mOld : tmax;
            aL[tid] = __expf(mOld - mNew);
            mI[tid] = mNew;
        }
        __syncthreads();
        // phase C - probabilities + segment sums
        {
            int row = tid >> 2, seg = tid & 3;
            float mNew = mI[row];
            float lsum = 0.f;
            #pragma unroll
            for (int j = 0; j < 16; ++j) {
                int c = seg * 16 + j;
                float p = __expf(sS[row * 65 + c] - mNew);
                sP[row * ALD + c] = f2bf(p);
                lsum += p;
            }
            sRed[tid] = lsum;
        }
        __syncthreads();
        // phase D - l update
        if (tid < 64) {
            float ls = sRed[tid * 4] + sRed[tid * 4 + 1] + sRed[tid * 4 + 2] + sRed[tid * 4 + 3];
            lI[tid] = lI[tid] * aL[tid] + ls;
        }
        __syncthreads();
        // rescale O, then O += P * V
        #pragma unroll
        for (int i = 0; i < 2; ++i) {
            int tile = wave * 2 + i, ms = tile >> 2, ns = tile & 3;
            int khalf = (lane >> 4) & 1;
            #pragma unroll
            for (int e = 0; e < 8; ++e)
                oacc[i][e] *= aL[ms * 16 + e + 8 * khalf];
            #pragma unroll
            for (int ks = 0; ks < 2; ++ks) {
                v16bf a  = frag_ld(sP,  ALD, ms * 16, ks * 32, lane);
                v16bf bb = frag_ld(sVt, ALD, ns * 16, ks * 32, lane);
                oacc[i] = wmma_bf16(a, bb, oacc[i]);
            }
        }
        __syncthreads();
    }
    // final normalize + bf16 store
    #pragma unroll
    for (int i = 0; i < 2; ++i) {
        int tile = wave * 2 + i, ms = tile >> 2, ns = tile & 3;
        int khalf = (lane >> 4) & 1, nn = ns * 16 + (lane & 15);
        #pragma unroll
        for (int e = 0; e < 8; ++e) {
            int row = ms * 16 + e + 8 * khalf;
            Out[((size_t)b * SS + qbase + row) * DD + hh * 64 + nn] =
                f2bf(oacc[i][e] / lI[row]);
        }
    }
}

// ---------------- tiny MLP head on token 1 ----------------
__global__ void __launch_bounds__(256)
head_kernel(const float* __restrict__ x,
            const float* __restrict__ l1w, const float* __restrict__ l1b,
            const float* __restrict__ l2w, const float* __restrict__ l2b,
            const float* __restrict__ l3w, const float* __restrict__ l3b,
            float* __restrict__ out) {
    __shared__ float e[2 * 768];
    __shared__ float h1[2 * 512];
    __shared__ float h2[2 * 250];
    int tid = threadIdx.x;
    for (int idx = tid; idx < 2 * 768; idx += 256) {
        int b = idx / 768, c = idx % 768;
        e[idx] = x[((size_t)b * SS + 1) * DD + c];
    }
    __syncthreads();
    for (int idx = tid; idx < 2 * 512; idx += 256) {
        int b = idx / 512, o = idx % 512;
        float s = l1b[o];
        for (int c = 0; c < 768; ++c) s += e[b * 768 + c] * l1w[(size_t)c * 512 + o];
        h1[idx] = s > 0.f ? s : 0.f;
    }
    __syncthreads();
    for (int idx = tid; idx < 2 * 250; idx += 256) {
        int b = idx / 250, o = idx % 250;
        float s = l2b[o];
        for (int c = 0; c < 512; ++c) s += h1[b * 512 + c] * l2w[(size_t)c * 250 + o];
        h2[idx] = s > 0.f ? s : 0.f;
    }
    __syncthreads();
    if (tid < 2) {
        float s = l3b[0];
        for (int c = 0; c < 250; ++c) s += h2[tid * 250 + c] * l3w[c];
        out[tid] = s;
        out[2 + tid] = s;
    }
}

// ---------------- driver ----------------
extern "C" void kernel_launch(void* const* d_in, const int* in_sizes, int n_in,
                              void* d_out, int out_size, void* d_ws, size_t ws_size,
                              hipStream_t stream) {
    const int*   ids   = (const int*)  d_in[0];
    const int*   amask = (const int*)  d_in[1];
    const float* wemb  = (const float*)d_in[2];
    const float* pemb  = (const float*)d_in[3];
    const float* elng  = (const float*)d_in[4];
    const float* elnb  = (const float*)d_in[5];
    const float* Wq    = (const float*)d_in[6];
    const float* bq    = (const float*)d_in[7];
    const float* Wk    = (const float*)d_in[8];
    const float* bk    = (const float*)d_in[9];
    const float* Wv    = (const float*)d_in[10];
    const float* bv    = (const float*)d_in[11];
    const float* Wo    = (const float*)d_in[12];
    const float* bo    = (const float*)d_in[13];
    const float* g1    = (const float*)d_in[14];
    const float* bn1   = (const float*)d_in[15];
    const float* W1    = (const float*)d_in[16];
    const float* b1f   = (const float*)d_in[17];
    const float* W2    = (const float*)d_in[18];
    const float* b2f   = (const float*)d_in[19];
    const float* g2    = (const float*)d_in[20];
    const float* bn2   = (const float*)d_in[21];
    const float* l1w   = (const float*)d_in[22];
    const float* l1b   = (const float*)d_in[23];
    const float* l2w   = (const float*)d_in[24];
    const float* l2b   = (const float*)d_in[25];
    const float* l3w   = (const float*)d_in[26];
    const float* l3b   = (const float*)d_in[27];

    const size_t ND  = (size_t)NTOK * DD;
    const size_t NF  = (size_t)NTOK * FFD;
    const size_t WDD = (size_t)LL * DD * DD;
    const size_t WDF = (size_t)LL * DD * FFD;

    float* ws   = (float*)d_ws;
    float* x    = ws;              // [NTOK, D] f32
    float* x1   = x  + ND;         // [NTOK, D] f32
    float* rbuf = x1 + ND;         // [NTOK, D] f32 residual branch
    __bf16* xbf  = (__bf16*)(rbuf + ND);
    __bf16* x1bf = xbf  + ND;
    __bf16* qbf  = x1bf + ND;
    __bf16* kbf  = qbf  + ND;
    __bf16* vtbf = kbf  + ND;      // V transposed [b][h][dh][s]
    __bf16* abf  = vtbf + ND;
    __bf16* hbf  = abf  + ND;      // [NTOK, FF]
    __bf16* wqb  = hbf + NF;       // weights, transposed bf16 [N][K]
    __bf16* wkb  = wqb + WDD;
    __bf16* wvb  = wkb + WDD;
    __bf16* wob  = wvb + WDD;
    __bf16* w1b  = wob + WDD;      // [FF][D] per layer
    __bf16* w2b  = w1b + WDF;      // [D][FF] per layer

    // one-shot weight convert+transpose
    {
        dim3 gdd(DD / 32,  DD / 32,  LL);
        dim3 g1d(DD / 32,  FFD / 32, LL);   // W1: K=D,  N=FF
        dim3 g2d(FFD / 32, DD / 32,  LL);   // W2: K=FF, N=D
        transpose_convert_kernel<<<gdd, 256, 0, stream>>>(Wq, wqb, DD, DD);
        transpose_convert_kernel<<<gdd, 256, 0, stream>>>(Wk, wkb, DD, DD);
        transpose_convert_kernel<<<gdd, 256, 0, stream>>>(Wv, wvb, DD, DD);
        transpose_convert_kernel<<<gdd, 256, 0, stream>>>(Wo, wob, DD, DD);
        transpose_convert_kernel<<<g1d, 256, 0, stream>>>(W1, w1b, DD, FFD);
        transpose_convert_kernel<<<g2d, 256, 0, stream>>>(W2, w2b, FFD, DD);
    }

    embed_ln_kernel<<<NTOK, 256, 0, stream>>>(ids, wemb, pemb, elng, elnb, x, xbf);

    dim3 gridD(NTOK / 128, DD / 128);    // (64, 6)
    dim3 gridF(NTOK / 128, FFD / 128);   // (64, 24)

    for (int l = 0; l < LL; ++l) {
        const __bf16* Wql = wqb + (size_t)l * DD * DD;
        const __bf16* Wkl = wkb + (size_t)l * DD * DD;
        const __bf16* Wvl = wvb + (size_t)l * DD * DD;
        const __bf16* Wol = wob + (size_t)l * DD * DD;
        const __bf16* W1l = w1b + (size_t)l * DD * FFD;
        const __bf16* W2l = w2b + (size_t)l * FFD * DD;

        gemm_bias_kernel<<<gridD, 256, 0, stream>>>(xbf, Wql, bq + (size_t)l * DD,
                                                    nullptr, qbf, nullptr, DD, DD, 0);
        gemm_bias_kernel<<<gridD, 256, 0, stream>>>(xbf, Wkl, bk + (size_t)l * DD,
                                                    nullptr, kbf, nullptr, DD, DD, 0);
        gemm_bias_kernel<<<gridD, 256, 0, stream>>>(xbf, Wvl, bv + (size_t)l * DD,
                                                    nullptr, nullptr, vtbf, DD, DD, 0);

        band_attn_kernel<<<BB * HH * NBK * 4, 256, 0, stream>>>(qbf, kbf, vtbf, amask, abf);

        gemm_bias_kernel<<<gridD, 256, 0, stream>>>(abf, Wol, bo + (size_t)l * DD,
                                                    rbuf, nullptr, nullptr, DD, DD, 0);
        add_ln_kernel<<<NTOK, 256, 0, stream>>>(x, rbuf, g1 + (size_t)l * DD,
                                                bn1 + (size_t)l * DD, x1, x1bf);

        gemm_bias_kernel<<<gridF, 256, 0, stream>>>(x1bf, W1l, b1f + (size_t)l * FFD,
                                                    nullptr, hbf, nullptr, FFD, DD, 1);
        gemm_bias_kernel<<<gridD, 256, 0, stream>>>(hbf, W2l, b2f + (size_t)l * DD,
                                                    rbuf, nullptr, nullptr, DD, FFD, 0);
        add_ln_kernel<<<NTOK, 256, 0, stream>>>(x1, rbuf, g2 + (size_t)l * DD,
                                                bn2 + (size_t)l * DD, x, xbf);
    }

    head_kernel<<<1, 256, 0, stream>>>(x, l1w, l1b, l2w, l2b, l3w, l3b, (float*)d_out);
}
